// MultiHeadSelfAttn_50465865728369
// MI455X (gfx1250) — matrix operand order repaired and checked by reading, not compiled, not run on hardware
//
#include <hip/hip_runtime.h>

typedef __bf16 bf16;
typedef bf16 v16bf __attribute__((ext_vector_type(16)));
typedef bf16 v8bf  __attribute__((ext_vector_type(8)));
typedef bf16 v4bf  __attribute__((ext_vector_type(4)));
typedef float v8f  __attribute__((ext_vector_type(8)));

#define T_SEQ 2048
#define NH    16
#define DH    64
#define CDIM  1024

// ---------------------------------------------------------------------------
// CDNA5 async global->LDS staging (ASYNCcnt-tracked, no VGPR round trip).
// lds_off = LDS byte address (low 32 bits of generic pointer to __shared__).
// ---------------------------------------------------------------------------
__device__ __forceinline__ void async_ld_b128(unsigned int lds_off, const void* gptr) {
  asm volatile("global_load_async_to_lds_b128 %0, %1, off"
               :: "v"(lds_off), "v"(gptr)
               : "memory");
}
__device__ __forceinline__ void wait_async0() {
  asm volatile("s_wait_asynccnt 0x0" ::: "memory");
}
__device__ __forceinline__ void wait_async4() {
  asm volatile("s_wait_asynccnt 0x4" ::: "memory");
}
__device__ __forceinline__ unsigned int lds_off32(const void* p) {
  return (unsigned int)(unsigned long long)(uintptr_t)p;
}

// ---------------------------------------------------------------------------
// WMMA fragment loads from LDS (wave32, v_wmma_f32_16x16x32_bf16 layouts)
// A 16x32: lane<16 holds row=lane, K {0..7,16..23}; lane>=16 same row, K+8.
// B 32x16: lane&15 = column, lanes 0-15 hold K {0..7,16..23}, lanes 16-31 K+8.
// ---------------------------------------------------------------------------
__device__ __forceinline__ v16bf frag_a(const bf16* p0, int stride) {
  const int lane = threadIdx.x & 31;
  const bf16* p = p0 + (lane & 15) * stride + ((lane >> 4) << 3);
  v8bf lo = *(const v8bf*)p;
  v8bf hi = *(const v8bf*)(p + 16);
  v16bf r;
#pragma unroll
  for (int i = 0; i < 8; ++i) { r[i] = lo[i]; r[i + 8] = hi[i]; }
  return r;
}

__device__ __forceinline__ v16bf frag_b(const bf16* p0, int stride) {
  const int lane = threadIdx.x & 31;
  const bf16* p = p0 + (lane & 15) * stride + ((lane >> 4) << 3);
  v8bf lo = *(const v8bf*)p;
  v8bf hi = *(const v8bf*)(p + 16);
  v16bf r;
#pragma unroll
  for (int i = 0; i < 8; ++i) { r[i] = lo[i]; r[i + 8] = hi[i]; }
  return r;
}

__device__ __forceinline__ v8f mma(v16bf a, v16bf b, v8f c) {
  return __builtin_amdgcn_wmma_f32_16x16x32_bf16(false, a, false, b,
                                                 (short)0, c, false, false);
}

// ---------------------------------------------------------------------------
// Prep: f32 -> bf16 convert, and transposed convert for weight matrices
// ---------------------------------------------------------------------------
__global__ void cvt_f32_bf16(const float* __restrict__ in, bf16* __restrict__ out, int n) {
  int i = blockIdx.x * blockDim.x + threadIdx.x;
  if (i < n) out[i] = (bf16)in[i];
}

// out[n*K + k] = (bf16) in[k*N + n]   (W is [K,N] row-major -> Wt [N,K])
__global__ void transpose_to_bf16(const float* __restrict__ in, bf16* __restrict__ out,
                                  int K, int N) {
  int i = blockIdx.x * blockDim.x + threadIdx.x;
  if (i < K * N) {
    int n = i / K, k = i % K;
    out[i] = (bf16)in[(size_t)k * N + n];
  }
}

// ---------------------------------------------------------------------------
// Per-K-step tile issue for the 128x128 GEMMs: 4 async b128 per thread.
// ---------------------------------------------------------------------------
__device__ __forceinline__ void issue_gemm_tile(
    const bf16* __restrict__ A, const bf16* __restrict__ Bt,
    int m0, int n0, int K, int k0,
    unsigned int aoff, unsigned int boff, int tid) {
#pragma unroll
  for (int c = 0; c < 2; ++c) {
    int chunk = c * 256 + tid;              // 512 chunks of 8 bf16 per matrix
    int row = chunk >> 2, kc = (chunk & 3) << 3;
    unsigned int lo = (unsigned int)(row * 40 + kc) * 2u;
    async_ld_b128(aoff + lo, &A[(size_t)(m0 + row) * K + k0 + kc]);
    async_ld_b128(boff + lo, &Bt[(size_t)(n0 + row) * K + k0 + kc]);
  }
}

// ---------------------------------------------------------------------------
// GEMM 1: qkv = x @ W_qkv + b_qkv ; epilogue scatters to Q[B,H,T,Dh],
// K[B,H,T,Dh], Vt[B,H,Dh,T] (bf16). 128x128 tile, 8 waves (2x4),
// double-buffered async-to-LDS staging.
// ---------------------------------------------------------------------------
__global__ __launch_bounds__(256) void gemm_qkv(
    const bf16* __restrict__ A, const bf16* __restrict__ Bt,
    const float* __restrict__ bias,
    bf16* __restrict__ qb, bf16* __restrict__ kb, bf16* __restrict__ vtb,
    int M, int N, int K) {
  __shared__ __align__(16) bf16 sA[2][128 * 40];
  __shared__ __align__(16) bf16 sB[2][128 * 40];
  const int tid = threadIdx.x, lane = tid & 31, wid = tid >> 5;
  const int wm = (wid & 1) * 64, wn = (wid >> 1) * 32;
  const int m0 = blockIdx.x * 128, n0 = blockIdx.y * 128;
  const unsigned int aoff[2] = { lds_off32(&sA[0][0]), lds_off32(&sA[1][0]) };
  const unsigned int boff[2] = { lds_off32(&sB[0][0]), lds_off32(&sB[1][0]) };

  v8f acc[4][2] = {};
  const int nk = K >> 5;

  issue_gemm_tile(A, Bt, m0, n0, K, 0, aoff[0], boff[0], tid);

  for (int ks = 0; ks < nk; ++ks) {
    const int cur = ks & 1;
    if (ks + 1 < nk) {
      issue_gemm_tile(A, Bt, m0, n0, K, (ks + 1) << 5,
                      aoff[cur ^ 1], boff[cur ^ 1], tid);
      wait_async4();                 // in-order: current tile's 4 are done
    } else {
      wait_async0();
    }
    __syncthreads();

    v16bf af[4], bfr[2];
#pragma unroll
    for (int mi = 0; mi < 4; ++mi) af[mi] = frag_a(&sA[cur][(wm + mi * 16) * 40], 40);
#pragma unroll
    for (int ni = 0; ni < 2; ++ni) bfr[ni] = frag_b(&sB[cur][(wn + ni * 16) * 40], 40);
#pragma unroll
    for (int mi = 0; mi < 4; ++mi)
#pragma unroll
      for (int ni = 0; ni < 2; ++ni)
        acc[mi][ni] = mma(af[mi], bfr[ni], acc[mi][ni]);
    __syncthreads();                 // all reads of 'cur' done before reuse
  }

  const int rbase = (lane >> 4) << 3, cbase = lane & 15;
#pragma unroll
  for (int mi = 0; mi < 4; ++mi)
#pragma unroll
    for (int ni = 0; ni < 2; ++ni)
#pragma unroll
      for (int i = 0; i < 8; ++i) {
        int m = m0 + wm + mi * 16 + rbase + i;
        int n = n0 + wn + ni * 16 + cbase;
        float v = acc[mi][ni][i] + bias[n];
        int sec = n >> 10, nn = n & (CDIM - 1);
        int h = nn >> 6, d = nn & (DH - 1);
        int b = m >> 11, t = m & (T_SEQ - 1);
        size_t bh = (size_t)(b * NH + h);
        if (sec == 0)      qb[(bh * T_SEQ + t) * DH + d] = (bf16)v;
        else if (sec == 1) kb[(bh * T_SEQ + t) * DH + d] = (bf16)v;
        else               vtb[(bh * DH + d) * T_SEQ + t] = (bf16)v;
      }
}

// ---------------------------------------------------------------------------
// GEMM 2: scores S = (Q Kt) * 1/sqrt(Dh), causal mask, lower-tri tiles only.
// grid.x = triangular tile index (136), grid.y = b*H+h. Async tile staging.
// ---------------------------------------------------------------------------
__global__ __launch_bounds__(256) void attn_scores(
    const bf16* __restrict__ qb, const bf16* __restrict__ kb,
    float* __restrict__ att) {
  __shared__ __align__(16) bf16 sQ[128 * 72];
  __shared__ __align__(16) bf16 sK[128 * 72];
  const int tid = threadIdx.x, lane = tid & 31, wid = tid >> 5;
  const int bh = blockIdx.y;
  int idx = blockIdx.x;
  int qt = 0;
  while ((qt + 1) * (qt + 2) / 2 <= idx) ++qt;     // triangular decode
  int kt = idx - qt * (qt + 1) / 2;

  const bf16* Q  = qb + (size_t)bh * T_SEQ * DH + (size_t)qt * 128 * DH;
  const bf16* Kp = kb + (size_t)bh * T_SEQ * DH + (size_t)kt * 128 * DH;
  const unsigned int qoff = lds_off32(&sQ[0]);
  const unsigned int koff = lds_off32(&sK[0]);

#pragma unroll
  for (int c = 0; c < 4; ++c) {
    int chunk = c * 256 + tid;                // 1024 chunks of 8 bf16 each
    int row = chunk >> 3, kc = (chunk & 7) << 3;
    unsigned int lo = (unsigned int)(row * 72 + kc) * 2u;
    async_ld_b128(qoff + lo, &Q[(size_t)row * DH + kc]);
    async_ld_b128(koff + lo, &Kp[(size_t)row * DH + kc]);
  }
  wait_async0();
  __syncthreads();

  const int wm = (wid & 1) * 64, wn = (wid >> 1) * 32;
  v8f acc[4][2] = {};
#pragma unroll
  for (int ks = 0; ks < 2; ++ks) {
    int k0 = ks * 32;
    v16bf af[4], bfr[2];
#pragma unroll
    for (int mi = 0; mi < 4; ++mi) af[mi] = frag_a(&sQ[(wm + mi * 16) * 72 + k0], 72);
#pragma unroll
    for (int ni = 0; ni < 2; ++ni) bfr[ni] = frag_b(&sK[(wn + ni * 16) * 72 + k0], 72);
#pragma unroll
    for (int mi = 0; mi < 4; ++mi)
#pragma unroll
      for (int ni = 0; ni < 2; ++ni)
        acc[mi][ni] = mma(af[mi], bfr[ni], acc[mi][ni]);
  }

  const int rbase = (lane >> 4) << 3, cbase = lane & 15;
#pragma unroll
  for (int mi = 0; mi < 4; ++mi)
#pragma unroll
    for (int ni = 0; ni < 2; ++ni)
#pragma unroll
      for (int i = 0; i < 8; ++i) {
        int qg = qt * 128 + wm + mi * 16 + rbase + i;
        int kg = kt * 128 + wn + ni * 16 + cbase;
        float v = (kg <= qg) ? acc[mi][ni][i] * 0.125f : -1.0e30f;
        att[((size_t)bh * T_SEQ + qg) * T_SEQ + kg] = v;
      }
}

// ---------------------------------------------------------------------------
// Softmax (causal, in place). One block per row; writes zeros above diagonal.
// ---------------------------------------------------------------------------
__global__ __launch_bounds__(256) void softmax_causal(float* __restrict__ att) {
  __shared__ float red[256];
  const int row = blockIdx.x;
  const int q = row & (T_SEQ - 1);
  float* p = att + (size_t)row * T_SEQ;
  const int tid = threadIdx.x;

  float mx = -3.0e38f;
  for (int j = tid; j <= q; j += 256) mx = fmaxf(mx, p[j]);
  red[tid] = mx; __syncthreads();
#pragma unroll
  for (int s = 128; s > 0; s >>= 1) {
    if (tid < s) red[tid] = fmaxf(red[tid], red[tid + s]);
    __syncthreads();
  }
  mx = red[0]; __syncthreads();

  float sum = 0.f;
  for (int j = tid; j <= q; j += 256) sum += __expf(p[j] - mx);
  red[tid] = sum; __syncthreads();
#pragma unroll
  for (int s = 128; s > 0; s >>= 1) {
    if (tid < s) red[tid] += red[tid + s];
    __syncthreads();
  }
  const float inv = 1.0f / red[0];

  for (int j = tid; j < T_SEQ; j += 256) {
    float v = 0.f;
    if (j <= q) v = __expf(p[j] - mx) * inv;
    p[j] = v;
  }
}

// ---------------------------------------------------------------------------
// GEMM 3: y = att @ V  (att f32 -> bf16 in VALU; Vt staged via async path).
// Block = 128 q-rows x 64 cols, k-loop only over causal range.
// Output written head-merged as bf16 [B,T,C].
// ---------------------------------------------------------------------------
__global__ __launch_bounds__(256) void attn_v(
    const float* __restrict__ att, const bf16* __restrict__ vtb,
    bf16* __restrict__ yb) {
  __shared__ __align__(16) bf16 sA[128 * 40];
  __shared__ __align__(16) bf16 sV[64 * 40];
  const int tid = threadIdx.x, lane = tid & 31, wid = tid >> 5;
  const int qt = blockIdx.x, bh = blockIdx.y;
  const int b = bh >> 4, h = bh & (NH - 1);
  const float* Ar = att + ((size_t)bh * T_SEQ + (size_t)qt * 128) * T_SEQ;
  const bf16* Vt = vtb + (size_t)bh * DH * T_SEQ;
  const unsigned int voff = lds_off32(&sV[0]);

  const int wm = (wid & 3) * 32, wn = (wid >> 2) * 32;
  v8f acc[2][2] = {};
  const int ksteps = (qt + 1) * 4;            // causal: k <= q tile range

  for (int ks = 0; ks < ksteps; ++ks) {
    int k0 = ks * 32;
    __syncthreads();
    {   // V tile 64x32: async global->LDS, one b128 per thread
      int row = tid >> 2, kc = (tid & 3) << 3;
      async_ld_b128(voff + (unsigned int)(row * 40 + kc) * 2u,
                    &Vt[(size_t)row * T_SEQ + k0 + kc]);
    }
#pragma unroll
    for (int c = 0; c < 4; ++c) {             // att tile 128x32 f32 -> bf16
      int chunk = c * 256 + tid;
      int row = chunk >> 3, kc = (chunk & 7) << 2;
      float4 v = *(const float4*)&Ar[(size_t)row * T_SEQ + k0 + kc];
      v4bf o;
      o[0] = (bf16)v.x; o[1] = (bf16)v.y; o[2] = (bf16)v.z; o[3] = (bf16)v.w;
      *(v4bf*)&sA[row * 40 + kc] = o;
    }
    if (ks + 1 < ksteps)
      __builtin_prefetch(&Ar[(size_t)(tid >> 1) * T_SEQ + k0 + 32], 0, 1);
    wait_async0();
    __syncthreads();

    v16bf af[2], bfr[2];
#pragma unroll
    for (int mi = 0; mi < 2; ++mi) af[mi] = frag_a(&sA[(wm + mi * 16) * 40], 40);
#pragma unroll
    for (int ni = 0; ni < 2; ++ni) bfr[ni] = frag_b(&sV[(wn + ni * 16) * 40], 40);
#pragma unroll
    for (int mi = 0; mi < 2; ++mi)
#pragma unroll
      for (int ni = 0; ni < 2; ++ni)
        acc[mi][ni] = mma(af[mi], bfr[ni], acc[mi][ni]);
  }

  const int rbase = (lane >> 4) << 3, cbase = lane & 15;
#pragma unroll
  for (int mi = 0; mi < 2; ++mi)
#pragma unroll
    for (int ni = 0; ni < 2; ++ni)
#pragma unroll
      for (int i = 0; i < 8; ++i) {
        int t = qt * 128 + wm + mi * 16 + rbase + i;
        int c = h * DH + wn + ni * 16 + cbase;
        yb[((size_t)(b * T_SEQ + t)) * CDIM + c] = (bf16)acc[mi][ni][i];
      }
}

// ---------------------------------------------------------------------------
// GEMM 4: out = y @ W_out + b_out (f32 into d_out), double-buffered async.
// ---------------------------------------------------------------------------
__global__ __launch_bounds__(256) void gemm_out(
    const bf16* __restrict__ A, const bf16* __restrict__ Bt,
    const float* __restrict__ bias, float* __restrict__ out,
    int M, int N, int K) {
  __shared__ __align__(16) bf16 sA[2][128 * 40];
  __shared__ __align__(16) bf16 sB[2][128 * 40];
  const int tid = threadIdx.x, lane = tid & 31, wid = tid >> 5;
  const int wm = (wid & 1) * 64, wn = (wid >> 1) * 32;
  const int m0 = blockIdx.x * 128, n0 = blockIdx.y * 128;
  const unsigned int aoff[2] = { lds_off32(&sA[0][0]), lds_off32(&sA[1][0]) };
  const unsigned int boff[2] = { lds_off32(&sB[0][0]), lds_off32(&sB[1][0]) };

  v8f acc[4][2] = {};
  const int nk = K >> 5;

  issue_gemm_tile(A, Bt, m0, n0, K, 0, aoff[0], boff[0], tid);

  for (int ks = 0; ks < nk; ++ks) {
    const int cur = ks & 1;
    if (ks + 1 < nk) {
      issue_gemm_tile(A, Bt, m0, n0, K, (ks + 1) << 5,
                      aoff[cur ^ 1], boff[cur ^ 1], tid);
      wait_async4();
    } else {
      wait_async0();
    }
    __syncthreads();

    v16bf af[4], bfr[2];
#pragma unroll
    for (int mi = 0; mi < 4; ++mi) af[mi] = frag_a(&sA[cur][(wm + mi * 16) * 40], 40);
#pragma unroll
    for (int ni = 0; ni < 2; ++ni) bfr[ni] = frag_b(&sB[cur][(wn + ni * 16) * 40], 40);
#pragma unroll
    for (int mi = 0; mi < 4; ++mi)
#pragma unroll
      for (int ni = 0; ni < 2; ++ni)
        acc[mi][ni] = mma(af[mi], bfr[ni], acc[mi][ni]);
    __syncthreads();
  }

  const int rbase = (lane >> 4) << 3, cbase = lane & 15;
#pragma unroll
  for (int mi = 0; mi < 4; ++mi)
#pragma unroll
    for (int ni = 0; ni < 2; ++ni)
#pragma unroll
      for (int i = 0; i < 8; ++i) {
        int m = m0 + wm + mi * 16 + rbase + i;
        int n = n0 + wn + ni * 16 + cbase;
        out[(size_t)m * N + n] = acc[mi][ni][i] + bias[n];
      }
}

// ---------------------------------------------------------------------------
extern "C" void kernel_launch(void* const* d_in, const int* in_sizes, int n_in,
                              void* d_out, int out_size, void* d_ws, size_t ws_size,
                              hipStream_t stream) {
  const float* x     = (const float*)d_in[0];
  const float* W_qkv = (const float*)d_in[1];
  const float* b_qkv = (const float*)d_in[2];
  const float* W_out = (const float*)d_in[3];
  const float* b_out = (const float*)d_in[4];

  float* out = (float*)d_out;                        // y: [2,2048,1024]
  float* att = out + (size_t)2 * T_SEQ * CDIM;       // att: [2,16,2048,2048]

  char* ws = (char*)d_ws;
  bf16* x_bf   = (bf16*)ws; ws += (size_t)4096 * 1024 * 2;
  bf16* wqkv_t = (bf16*)ws; ws += (size_t)3072 * 1024 * 2;
  bf16* wout_t = (bf16*)ws; ws += (size_t)1024 * 1024 * 2;
  bf16* q_bf   = (bf16*)ws; ws += (size_t)2 * NH * T_SEQ * DH * 2;
  bf16* k_bf   = (bf16*)ws; ws += (size_t)2 * NH * T_SEQ * DH * 2;
  bf16* vt_bf  = (bf16*)ws; ws += (size_t)2 * NH * T_SEQ * DH * 2;
  bf16* y_bf   = (bf16*)ws;

  // Prep: bf16 conversions (+ weight transposes to [N,K])
  cvt_f32_bf16<<<(4194304 + 255) / 256, 256, 0, stream>>>(x, x_bf, 4194304);
  transpose_to_bf16<<<(3145728 + 255) / 256, 256, 0, stream>>>(W_qkv, wqkv_t, 1024, 3072);
  transpose_to_bf16<<<(1048576 + 255) / 256, 256, 0, stream>>>(W_out, wout_t, 1024, 1024);

  // 1) QKV projection -> Q,K head-major bf16, V head-transposed bf16
  gemm_qkv<<<dim3(32, 24), 256, 0, stream>>>(x_bf, wqkv_t, b_qkv,
                                             q_bf, k_bf, vt_bf, 4096, 3072, 1024);
  // 2) causal scores (lower-triangular tiles only)
  attn_scores<<<dim3(136, 32), 256, 0, stream>>>(q_bf, k_bf, att);
  // 3) softmax in place (also zero-fills above diagonal)
  softmax_causal<<<65536, 256, 0, stream>>>(att);
  // 4) att @ V -> merged-head bf16 activations
  attn_v<<<dim3(16, 32), 256, 0, stream>>>(att, vt_bf, y_bf);
  // 5) output projection -> f32 d_out
  gemm_out<<<dim3(32, 8), 256, 0, stream>>>(y_bf, wout_t, b_out, out, 4096, 1024, 1024);
}
